// Decoder_lstm_41824391528720
// MI455X (gfx1250) — compile-verified, matrix operand order checked by reading, not run on hardware
//
#include <hip/hip_runtime.h>

// ---------------------------------------------------------------------------
// Autoregressive LSTM decoder (B=64, T=512, L=256, H=1024) for gfx1250.
// Persistent-kernel recurrence, bf16 WMMA (v_wmma_f32_16x16x32_bf16),
// fp32 accumulate, c-state resident in LDS, grid barriers between phases.
// ---------------------------------------------------------------------------

#define BB 64
#define TT 512
#define LL 256
#define HH 1024
#define G4 4096       // 4*H
#define NWG 32
#define HS  32        // hidden slice per WG = H / NWG
#define WAVES 8

typedef unsigned short u16;
typedef __attribute__((ext_vector_type(16))) __bf16 v16bf;
typedef __attribute__((ext_vector_type(8)))  float  v8f;

__device__ __forceinline__ u16 f2bf(float f) {
    unsigned int u = __float_as_uint(f);
    u += 0x7FFFu + ((u >> 16) & 1u);          // round-to-nearest-even
    return (u16)(u >> 16);
}

__device__ __forceinline__ v16bf load_frag(const u16* p) {
    union { uint4 u[2]; v16bf v; } f;
    const uint4* q = (const uint4*)p;          // 32B-aligned by construction
    f.u[0] = q[0];
    f.u[1] = q[1];
    return f.v;
}

__device__ __forceinline__ float sigmoid_(float x) {
    return 1.0f / (1.0f + __expf(-x));
}
__device__ __forceinline__ float tanh_(float x) {
    return 1.0f - 2.0f / (__expf(2.0f * x) + 1.0f);
}

// ---------------------------------------------------------------------------
// Setup: fp32 -> bf16 weight conversion, bias fusion, state init, barrier init
// ---------------------------------------------------------------------------
__global__ void lstm_setup_kernel(const float* __restrict__ w_ih,
                                  const float* __restrict__ w_hh,
                                  const float* __restrict__ w_fc,
                                  const float* __restrict__ b_ih,
                                  const float* __restrict__ b_hh,
                                  const float* __restrict__ h0,
                                  const float* __restrict__ x0,
                                  u16* __restrict__ w_ih_bf,
                                  u16* __restrict__ w_hh_bf,
                                  u16* __restrict__ w_fc_bf,
                                  float* __restrict__ bias,
                                  u16* __restrict__ h_bf0,
                                  u16* __restrict__ x_bf0,
                                  int* __restrict__ bar) {
    long i = (long)blockIdx.x * blockDim.x + threadIdx.x;
    long stride = (long)gridDim.x * blockDim.x;
    for (long k = i; k < (long)G4 * HH; k += stride) w_hh_bf[k] = f2bf(w_hh[k]);
    for (long k = i; k < (long)G4 * LL; k += stride) w_ih_bf[k] = f2bf(w_ih[k]);
    for (long k = i; k < (long)LL * HH; k += stride) w_fc_bf[k] = f2bf(w_fc[k]);
    for (long k = i; k < G4;            k += stride) bias[k]    = b_ih[k] + b_hh[k];
    for (long k = i; k < BB * HH;       k += stride) h_bf0[k]   = f2bf(h0[k]);
    for (long k = i; k < BB * LL;       k += stride) x_bf0[k]   = f2bf(x0[k]);
    if (i < 2) bar[i] = 0;
}

// ---------------------------------------------------------------------------
// Grid barrier (all NWG workgroups resident; generation-counter spin)
// ---------------------------------------------------------------------------
__device__ __forceinline__ void grid_barrier(int* cnt, volatile int* gen, int epoch) {
    __syncthreads();
    if (threadIdx.x == 0) {
        __threadfence();
        int prev = atomicAdd(cnt, 1);
        if (prev == NWG - 1) {
            *cnt = 0;
            __threadfence();
            *gen = epoch;
        } else {
            while (*gen < epoch) __builtin_amdgcn_s_sleep(1);
        }
    }
    __syncthreads();
    __threadfence();
}

// ---------------------------------------------------------------------------
// Persistent recurrent kernel: 32 WGs x 256 threads (8 waves of 32)
// ---------------------------------------------------------------------------
__launch_bounds__(256, 1)
__global__ void lstm_decoder_kernel(const u16*  __restrict__ w_ih_bf,
                                    const u16*  __restrict__ w_hh_bf,
                                    const u16*  __restrict__ w_fc_bf,
                                    const float* __restrict__ bias,
                                    const float* __restrict__ b_fc,
                                    const float* __restrict__ c0,
                                    u16*  __restrict__ h_bf,   // 2 x [B*H]
                                    u16*  __restrict__ x_bf,   // 2 x [B*L]
                                    float* __restrict__ out,   // [B,1,T,L]
                                    int*  __restrict__ bar) {
    __shared__ float c_lds[BB * HS];           // persistent cell state slice

    const int wg    = blockIdx.x;
    const int tid   = threadIdx.x;
    const int lane  = tid & 31;
    const int wv    = tid >> 5;                // wave id 0..7 (wave32)
    const int l15   = lane & 15;
    const int khalf = (lane >> 4) * 16;        // K sub-offset: contiguous k = 16*half + s (same for A and B)
    const int slice = wg * HS;                 // hidden-unit slice base

    // --- load this WG's cell-state slice into LDS ---
    for (int idx = tid; idx < BB * HS; idx += 256) {
        int b = idx >> 5, j = idx & (HS - 1);
        c_lds[idx] = c0[b * HH + slice + j];
    }
    __syncthreads();

    int* cnt = bar;
    volatile int* gen = bar + 1;
    int epoch = 0;

    // phase-1 tile ownership: wave -> (batch tile, local j tile)
    const int mtile = wv >> 1;                 // 0..3  (batch 16-row tile)
    const int jt    = wv & 1;                  // 0..1  (16-wide j tile in slice)
    const int brow  = mtile * 16 + l15;        // A-matrix row (batch index)
    const int n_j   = slice + jt * 16 + l15;   // hidden index j for B/C columns

    // phase-2 tile ownership
    const int gwave = wg * WAVES + wv;

    for (int t = 0; t < TT; ++t) {
        const u16* x_cur = x_bf + (size_t)(t & 1) * BB * LL;
        const u16* h_cur = h_bf + (size_t)(t & 1) * BB * HH;
        u16* x_nxt = x_bf + (size_t)((t + 1) & 1) * BB * LL;
        u16* h_nxt = h_bf + (size_t)((t + 1) & 1) * BB * HH;

        // ---------------- Phase 1: gates + cell update ----------------
        v8f acc[4];
        v8f zero = {0.f, 0.f, 0.f, 0.f, 0.f, 0.f, 0.f, 0.f};
        acc[0] = zero; acc[1] = zero; acc[2] = zero; acc[3] = zero;

        // x @ W_ih^T contribution: K = 256 -> 8 k-tiles of 32
        for (int kt = 0; kt < 8; ++kt) {
            const int ko = kt * 32 + khalf;
            v16bf a = load_frag(x_cur + brow * LL + ko);
#pragma unroll
            for (int g = 0; g < 4; ++g) {
                const u16* wp = w_ih_bf + (size_t)(g * HH + n_j) * LL + ko;
                v16bf bf = load_frag(wp);
                acc[g] = __builtin_amdgcn_wmma_f32_16x16x32_bf16(
                    false, a, false, bf, (short)0, acc[g], false, false);
            }
        }
        // h @ W_hh^T contribution: K = 1024 -> 32 k-tiles of 32
        for (int kt = 0; kt < 32; ++kt) {
            const int ko = kt * 32 + khalf;
            v16bf a = load_frag(h_cur + brow * HH + ko);
            if (kt < 31)  // stream next k-tile of W_hh through L2
                __builtin_prefetch(w_hh_bf + (size_t)n_j * HH + ko + 32, 0, 1);
#pragma unroll
            for (int g = 0; g < 4; ++g) {
                const u16* wp = w_hh_bf + (size_t)(g * HH + n_j) * HH + ko;
                v16bf bf = load_frag(wp);
                acc[g] = __builtin_amdgcn_wmma_f32_16x16x32_bf16(
                    false, a, false, bf, (short)0, acc[g], false, false);
            }
        }

        // bias + activations + cell update (i,f,g,o all in this wave)
        const float bi = bias[0 * HH + n_j];
        const float bf_ = bias[1 * HH + n_j];
        const float bg = bias[2 * HH + n_j];
        const float bo = bias[3 * HH + n_j];
#pragma unroll
        for (int v = 0; v < 8; ++v) {
            const int b    = mtile * 16 + v + 8 * (lane >> 4);
            const int cidx = b * HS + jt * 16 + l15;
            float iv = sigmoid_(acc[0][v] + bi);
            float fv = sigmoid_(acc[1][v] + bf_);
            float gv = tanh_(acc[2][v] + bg);
            float ov = sigmoid_(acc[3][v] + bo);
            float cv = fv * c_lds[cidx] + iv * gv;
            c_lds[cidx] = cv;
            float hv = ov * tanh_(cv);
            h_nxt[(size_t)b * HH + n_j] = f2bf(hv);
        }

        grid_barrier(cnt, gen, ++epoch);       // h_nxt globally visible

        // ---------------- Phase 2: out = h_new @ W_fc^T + b_fc ----------------
        if (gwave < 64) {
            const int m2  = gwave >> 4;        // 0..3 batch tile
            const int n2t = gwave & 15;        // 0..15 output tile
            const int b2  = m2 * 16 + l15;
            const int nn  = n2t * 16 + l15;
            v8f acc2 = zero;
            for (int kt = 0; kt < 32; ++kt) {
                const int ko = kt * 32 + khalf;
                v16bf a = load_frag(h_nxt + (size_t)b2 * HH + ko);
                v16bf w = load_frag(w_fc_bf + (size_t)nn * HH + ko);
                acc2 = __builtin_amdgcn_wmma_f32_16x16x32_bf16(
                    false, a, false, w, (short)0, acc2, false, false);
            }
            const float bb = b_fc[nn];
#pragma unroll
            for (int v = 0; v < 8; ++v) {
                const int b = m2 * 16 + v + 8 * (lane >> 4);
                float val = acc2[v] + bb;
                out[((size_t)b * TT + t) * LL + nn] = val;   // [B,1,T,L]
                x_nxt[(size_t)b * LL + nn] = f2bf(val);      // autoregressive feed
            }
        }

        grid_barrier(cnt, gen, ++epoch);       // x_nxt globally visible
    }
}

// ---------------------------------------------------------------------------
// Host-side launch (graph-capture safe: only kernel launches on `stream`)
// ---------------------------------------------------------------------------
extern "C" void kernel_launch(void* const* d_in, const int* in_sizes, int n_in,
                              void* d_out, int out_size, void* d_ws, size_t ws_size,
                              hipStream_t stream) {
    const float* x0   = (const float*)d_in[0];   // initial_input [B,L]
    // d_in[1] encoder_outputs: unused by forward
    const float* h0   = (const float*)d_in[2];   // [1,B,H]
    const float* c0   = (const float*)d_in[3];   // [1,B,H]
    // d_in[4] targets, d_in[5] tf_prob: tf_prob==0 -> always autoregressive
    const float* w_ih = (const float*)d_in[6];   // [4H,L]
    const float* w_hh = (const float*)d_in[7];   // [4H,H]
    const float* b_ih = (const float*)d_in[8];   // [4H]
    const float* b_hh = (const float*)d_in[9];   // [4H]
    const float* w_fc = (const float*)d_in[10];  // [L,H]
    const float* b_fc = (const float*)d_in[11];  // [L]

    // workspace carve-up (all sizes multiples of 32B)
    char* ws = (char*)d_ws;
    u16* w_hh_bf = (u16*)ws; ws += (size_t)G4 * HH * sizeof(u16);   // 8 MB
    u16* w_ih_bf = (u16*)ws; ws += (size_t)G4 * LL * sizeof(u16);   // 2 MB
    u16* w_fc_bf = (u16*)ws; ws += (size_t)LL * HH * sizeof(u16);   // 0.5 MB
    float* bias  = (float*)ws; ws += (size_t)G4 * sizeof(float);    // 16 KB
    u16* h_bf    = (u16*)ws; ws += (size_t)2 * BB * HH * sizeof(u16); // 256 KB
    u16* x_bf    = (u16*)ws; ws += (size_t)2 * BB * LL * sizeof(u16); // 64 KB
    int* bar     = (int*)ws;                                          // 2 ints

    lstm_setup_kernel<<<2048, 256, 0, stream>>>(
        w_ih, w_hh, w_fc, b_ih, b_hh, h0, x0,
        w_ih_bf, w_hh_bf, w_fc_bf, bias, h_bf, x_bf, bar);

    lstm_decoder_kernel<<<NWG, 256, 0, stream>>>(
        w_ih_bf, w_hh_bf, w_fc_bf, bias, b_fc, c0,
        h_bf, x_bf, (float*)d_out, bar);
}
